// GEM1Encoder_21414706938218
// MI455X (gfx1250) — compile-verified, matrix operand order checked by reading, not run on hardware
//
#include <hip/hip_runtime.h>
#include <hip/hip_bf16.h>

#define EPSN 1e-5f

typedef __attribute__((ext_vector_type(16))) __bf16 v16bf;
typedef __attribute__((ext_vector_type(8)))  float  v8f;

// ---------------------------------------------------------------- utilities
__global__ void zero_f32(float* __restrict__ p, size_t n) {
  size_t t = (size_t)blockIdx.x * blockDim.x + threadIdx.x;
  if (t < n) p[t] = 0.f;
}

__global__ void copy_f32(const float* __restrict__ s, float* __restrict__ d, size_t n) {
  size_t t = (size_t)blockIdx.x * blockDim.x + threadIdx.x;
  if (t < n) d[t] = s[t];
}

__global__ void count_batch(const int* __restrict__ b, int n, float* __restrict__ cnt) {
  int t = blockIdx.x * blockDim.x + threadIdx.x;
  if (t < n) atomicAdd(&cnt[b[t]], 1.f);
}

// ------------------------------------------------- weight swizzle for WMMA B
// W is row-major [NI x NO] (input-dim rows). Output: for each 16-col tile ct,
// 32-K chunk kc, lane, 16 contiguous bf16 in exact B-fragment order.
__global__ void swizzle_w(const float* __restrict__ W, int NI, int NO,
                          __bf16* __restrict__ wsw) {
  int t = blockIdx.x * blockDim.x + threadIdx.x;
  int total = NI * NO;
  if (t >= total) return;
  int e    = t & 15;
  int lane = (t >> 4) & 31;
  int rest = t >> 9;             // ct*nchunks + kc
  int nchunks = NI >> 5;
  int ct = rest / nchunks, kc = rest - ct * nchunks;
  int v = e >> 1, j = e & 1;
  int K = kc * 32 + 2 * v + j + ((v >= 4) ? 8 : 0) + ((lane >= 16) ? 8 : 0);
  int Nc = ct * 16 + (lane & 15);
  wsw[t] = (__bf16)W[(size_t)K * NO + Nc];
}

// -------------------------------------------------------------- WMMA GEMM
// OUT[R x NO] = TR(IN[R x NI]) @ W + bias ; TR: mode 0 identity, 1 affine+relu
// block = 128 threads (4 waves); grid = ceil(R/16). A tile staged into LDS
// ONCE per 16 rows; each wave loops over its share of 16-col output tiles so
// activation HBM traffic is minimal (weights are L2-resident).
__global__ __launch_bounds__(128)
void gemm_wmma(const float* __restrict__ in, int R, int NI, int NO,
               const __bf16* __restrict__ wsw, const float* __restrict__ bias,
               const float* __restrict__ colA, const float* __restrict__ colB,
               int mode, float* __restrict__ out) {
  __shared__ __bf16 Asl[16 * 256];
  int row0 = blockIdx.x * 16;
  int wave = threadIdx.x >> 5;
  int lane = threadIdx.x & 31;

  for (int idx = threadIdx.x; idx < 16 * NI; idx += 128) {
    int r = idx / NI, k = idx - r * NI;
    int rr = row0 + r;
    float v = 0.f;
    if (rr < R) v = in[(size_t)rr * NI + k];
    if (mode == 1) { v = v * colA[k] + colB[k]; v = v > 0.f ? v : 0.f; }
    Asl[r * NI + k] = (__bf16)v;
  }
  __syncthreads();

  int nchunks = NI >> 5;
  int nct = NO >> 4;
  int Ml = lane & 15;
  int hi = (lane >> 4) << 3;               // 0 or 8
  int rbase = row0 + ((lane >= 16) ? 8 : 0);

  for (int ct = wave; ct < nct; ct += 4) {
    v8f acc = {0.f,0.f,0.f,0.f,0.f,0.f,0.f,0.f};
    const __bf16* wp = wsw + ((size_t)(ct * nchunks) * 32 + lane) * 16;
    for (int kc = 0; kc < nchunks; ++kc) {
      v16bf b = *((const v16bf*)(wp + (size_t)kc * 512));
      v16bf a;
      int kb = kc * 32;
#pragma unroll
      for (int e = 0; e < 16; ++e) {
        int v = e >> 1, j = e & 1;
        int K = kb + 2 * v + j + ((v >= 4) ? 8 : 0) + hi;
        a[e] = Asl[Ml * NI + K];
      }
      acc = __builtin_amdgcn_wmma_f32_16x16x32_bf16(false, a, false, b,
                                                    (short)0, acc, false, false);
    }
    int col = ct * 16 + Ml;
    float bb = bias[col];
#pragma unroll
    for (int r = 0; r < 8; ++r) {
      int row = rbase + r;
      if (row < R) out[(size_t)row * NO + col] = acc[r] + bb;
    }
  }
}

// ---------------------------------------------------------------- encoders
struct Tab9 { const float* t[9]; };
struct Tab3 { const float* t[3]; };

__global__ __launch_bounds__(128)
void enc_atom(Tab9 tabs, const int* __restrict__ xi,
              const float* __restrict__ pg, const float* __restrict__ pe,
              const float* __restrict__ Wd, const float* __restrict__ bd,
              const float* __restrict__ Wp, const float* __restrict__ bp,
              int N, float* __restrict__ outx) {
  int i = blockIdx.x; if (i >= N) return;
  int f = threadIdx.x;
  __shared__ float feat[256];
  __shared__ float rr[30];
  float dx = pg[i*3+0] - pe[i*3+0] + 1e-6f;
  float dy = pg[i*3+1] - pe[i*3+1] + 1e-6f;
  float dz = pg[i*3+2] - pe[i*3+2] + 1e-6f;
  float d = sqrtf(dx*dx + dy*dy + dz*dz);
  if (f < 30) { float u = d - 0.1f * f; rr[f] = __expf(-10.f * u * u); }
  float emb = 0.f;
#pragma unroll
  for (int t = 0; t < 9; ++t) emb += tabs.t[t][(size_t)xi[i*9+t] * 128 + f];
  __syncthreads();
  float rb = bd[f];
  for (int k = 0; k < 30; ++k) rb += rr[k] * Wd[k*128 + f];
  feat[f] = emb; feat[128+f] = rb;
  __syncthreads();
  float o = bp[f];
  for (int k = 0; k < 256; ++k) o += feat[k] * Wp[k*128 + f];
  outx[(size_t)i*128 + f] = o;
}

__global__ __launch_bounds__(128)
void enc_bond(Tab3 tabs, const int* __restrict__ ea,
              const float* __restrict__ lg, const float* __restrict__ le,
              const float* __restrict__ Wl, const float* __restrict__ bl,
              const float* __restrict__ Wp, const float* __restrict__ bp,
              int E, float* __restrict__ oute) {
  int i = blockIdx.x; if (i >= E) return;
  int f = threadIdx.x;
  __shared__ float feat[256];
  __shared__ float r1[20], r2[20];
  if (f < 20) {
    float u = lg[i] - 0.1f * f; r1[f] = __expf(-10.f * u * u);
    float v = le[i] - 0.1f * f; r2[f] = __expf(-10.f * v * v);
  }
  float emb = 0.f;
#pragma unroll
  for (int t = 0; t < 3; ++t) emb += tabs.t[t][(size_t)ea[i*3+t] * 128 + f];
  __syncthreads();
  float a = bl[f], b = bl[f];
  for (int k = 0; k < 20; ++k) { a += r1[k]*Wl[k*128+f]; b += r2[k]*Wl[k*128+f]; }
  feat[f] = a; feat[128+f] = b;
  __syncthreads();
  float o = bp[f];
  for (int k = 0; k < 256; ++k) o += feat[k] * Wp[k*128 + f];
  oute[(size_t)i*128 + f] = emb + o;
}

__global__ __launch_bounds__(128)
void enc_angle(const float* __restrict__ ag, const float* __restrict__ ae,
               const float* __restrict__ Wa, const float* __restrict__ ba,
               const float* __restrict__ Wp, const float* __restrict__ bp,
               int A, float* __restrict__ outa) {
  int i = blockIdx.x; if (i >= A) return;
  int f = threadIdx.x;
  __shared__ float feat[256];
  __shared__ float r1[32], r2[32];
  if (f < 32) {
    float u = ag[i] - 0.1f * f; r1[f] = __expf(-10.f * u * u);
    float v = ae[i] - 0.1f * f; r2[f] = __expf(-10.f * v * v);
  }
  __syncthreads();
  float a = ba[f], b = ba[f];
  for (int k = 0; k < 32; ++k) { a += r1[k]*Wa[k*128+f]; b += r2[k]*Wa[k*128+f]; }
  feat[f] = a; feat[128+f] = b;
  __syncthreads();
  float o = bp[f];
  for (int k = 0; k < 256; ++k) o += feat[k] * Wp[k*128 + f];
  outa[(size_t)i*128 + f] = o;
}

// --------------------------------------------------------- GIN block pieces
__global__ void gin_messages(const float* __restrict__ x, const float* __restrict__ ea,
                             const int* __restrict__ srcv, const int* __restrict__ dstv,
                             int ne, float* __restrict__ h0) {
  size_t t = (size_t)blockIdx.x * blockDim.x + threadIdx.x;
  size_t total = (size_t)ne * 32;
  if (t >= total) return;
  int e = (int)(t >> 5);
  int q = (int)(t & 31) * 4;
  int s = srcv[e], d = dstv[e];
  float4 xv = *(const float4*)(x + (size_t)s*128 + q);
  float4 ev = *(const float4*)(ea + (size_t)e*128 + q);
  float m0 = fmaxf(xv.x + ev.x, 0.f), m1 = fmaxf(xv.y + ev.y, 0.f);
  float m2 = fmaxf(xv.z + ev.z, 0.f), m3 = fmaxf(xv.w + ev.w, 0.f);
  float* p = h0 + (size_t)d*128 + q;
  atomicAdd(p+0, m0); atomicAdd(p+1, m1); atomicAdd(p+2, m2); atomicAdd(p+3, m3);
}

__global__ __launch_bounds__(256)
void bn_stats(const float* __restrict__ h, int R, float* __restrict__ s, float* __restrict__ q) {
  int col = threadIdx.x;
  int r0 = blockIdx.x * 256;
  int r1 = r0 + 256; if (r1 > R) r1 = R;
  float S = 0.f, Q = 0.f;
  for (int r = r0; r < r1; ++r) { float v = h[(size_t)r*256 + col]; S += v; Q += v*v; }
  atomicAdd(&s[col], S); atomicAdd(&q[col], Q);
}

__global__ void bn_finalize(const float* __restrict__ s, const float* __restrict__ q,
                            const float* __restrict__ g, const float* __restrict__ b,
                            int R, float* __restrict__ a, float* __restrict__ bo) {
  int c = threadIdx.x;
  float inv = 1.f / (float)R;
  float mu = s[c] * inv;
  float var = q[c] * inv - mu * mu;
  float aa = g[c] * rsqrtf(var + EPSN);
  a[c] = aa; bo[c] = b[c] - mu * aa;
}

__global__ __launch_bounds__(128)
void ln_rowstats(const float* __restrict__ h, const int* __restrict__ gb, int R,
                 float* __restrict__ gsum, float* __restrict__ gsq) {
  int row = blockIdx.x; if (row >= R) return;
  float v = h[(size_t)row*128 + threadIdx.x];
  float s = v, q = v * v;
  for (int o = 16; o > 0; o >>= 1) { s += __shfl_down(s, o, 32); q += __shfl_down(q, o, 32); }
  __shared__ float ss[4], qq[4];
  if ((threadIdx.x & 31) == 0) { ss[threadIdx.x >> 5] = s; qq[threadIdx.x >> 5] = q; }
  __syncthreads();
  if (threadIdx.x == 0) {
    float S = ss[0]+ss[1]+ss[2]+ss[3];
    float Q = qq[0]+qq[1]+qq[2]+qq[3];
    int g = gb[row];
    atomicAdd(&gsum[g], S); atomicAdd(&gsq[g], Q);
  }
}

__global__ void ln_finalize(const float* __restrict__ gsum, const float* __restrict__ gsq,
                            const float* __restrict__ cnt, int G,
                            float* __restrict__ gmean, float* __restrict__ ginv) {
  int g = blockIdx.x * blockDim.x + threadIdx.x; if (g >= G) return;
  float c = fmaxf(cnt[g], 1.f) * 128.f;
  float m = gsum[g] / c;
  float var = gsq[g] / c - m * m;
  gmean[g] = m; ginv[g] = rsqrtf(var + EPSN);
}

__global__ void ln_apply(float* __restrict__ h, const int* __restrict__ gb, int R,
                         const float* __restrict__ gmean, const float* __restrict__ ginv,
                         const float* __restrict__ lnw, const float* __restrict__ lnb,
                         float* __restrict__ gmf) {
  size_t t = (size_t)blockIdx.x * blockDim.x + threadIdx.x;
  if (t >= (size_t)R * 128) return;
  int row = (int)(t >> 7), f = (int)(t & 127);
  int g = gb[row];
  float v = (h[t] - gmean[g]) * ginv[g] * lnw[f] + lnb[f];
  h[t] = v;
  atomicAdd(&gmf[(size_t)g * 128 + f], v);
}

__global__ void gn_div(float* __restrict__ gmf, const float* __restrict__ cnt, int G) {
  size_t t = (size_t)blockIdx.x * blockDim.x + threadIdx.x;
  if (t >= (size_t)G * 128) return;
  gmf[t] /= fmaxf(cnt[t >> 7], 1.f);
}

__global__ void gn_apply1(float* __restrict__ h, const int* __restrict__ gb, int R,
                          const float* __restrict__ gmf, const float* __restrict__ gnms,
                          float* __restrict__ gvf) {
  size_t t = (size_t)blockIdx.x * blockDim.x + threadIdx.x;
  if (t >= (size_t)R * 128) return;
  int row = (int)(t >> 7), f = (int)(t & 127);
  int g = gb[row];
  float o = h[t] - gmf[(size_t)g * 128 + f] * gnms[f];
  h[t] = o;
  atomicAdd(&gvf[(size_t)g * 128 + f], o * o);
}

__global__ void gn_finalize(float* __restrict__ gvf, const float* __restrict__ cnt, int G) {
  size_t t = (size_t)blockIdx.x * blockDim.x + threadIdx.x;
  if (t >= (size_t)G * 128) return;
  gvf[t] = rsqrtf(gvf[t] / fmaxf(cnt[t >> 7], 1.f) + EPSN);
}

__global__ void gn_apply2(const float* __restrict__ h, const int* __restrict__ gb, int R,
                          const float* __restrict__ gvf,
                          const float* __restrict__ gnw, const float* __restrict__ gnb,
                          int last_act, float* __restrict__ x) {
  size_t t = (size_t)blockIdx.x * blockDim.x + threadIdx.x;
  if (t >= (size_t)R * 128) return;
  int row = (int)(t >> 7), f = (int)(t & 127);
  int g = gb[row];
  float o = gnw[f] * h[t] * gvf[(size_t)g * 128 + f] + gnb[f];
  if (last_act) o = fmaxf(o, 0.f);
  x[t] += o;
}

// --------------------------------------------------------------- pool/final
__global__ void pool_add(const float* __restrict__ x, const int* __restrict__ batch,
                         int N, float* __restrict__ pool) {
  size_t t = (size_t)blockIdx.x * blockDim.x + threadIdx.x;
  if (t >= (size_t)N * 128) return;
  int row = (int)(t >> 7), f = (int)(t & 127);
  atomicAdd(&pool[(size_t)batch[row] * 128 + f], x[t]);
}

__global__ __launch_bounds__(128)
void final_ln(const float* __restrict__ pool, const float* __restrict__ cnt,
              const float* __restrict__ w, const float* __restrict__ b,
              int G, float* __restrict__ out) {
  int g = blockIdx.x; if (g >= G) return;
  int f = threadIdx.x;
  float c = fmaxf(cnt[g], 1.f);
  float v = pool[(size_t)g * 128 + f] / c;
  __shared__ float sbuf[128];
  __shared__ float mu_s, inv_s;
  sbuf[f] = v; __syncthreads();
  for (int o = 64; o > 0; o >>= 1) { if (f < o) sbuf[f] += sbuf[f + o]; __syncthreads(); }
  if (f == 0) mu_s = sbuf[0] * (1.f / 128.f);
  __syncthreads();
  float dv = v - mu_s;
  sbuf[f] = dv * dv; __syncthreads();
  for (int o = 64; o > 0; o >>= 1) { if (f < o) sbuf[f] += sbuf[f + o]; __syncthreads(); }
  if (f == 0) inv_s = rsqrtf(sbuf[0] * (1.f / 128.f) + EPSN);
  __syncthreads();
  out[(size_t)g * 128 + f] = dv * inv_s * w[f] + b[f];
}

// ================================================================ host side
struct GinParams {
  const float *W1,*b1,*bng,*bnb,*W2,*b2,*lnw,*lnb,*gnw,*gnb,*gnms;
};
static GinParams getGin(void* const* d_in, int base) {
  GinParams g;
  g.W1  = (const float*)d_in[base+0];  g.b1  = (const float*)d_in[base+1];
  g.bng = (const float*)d_in[base+2];  g.bnb = (const float*)d_in[base+3];
  g.W2  = (const float*)d_in[base+4];  g.b2  = (const float*)d_in[base+5];
  g.lnw = (const float*)d_in[base+6];  g.lnb = (const float*)d_in[base+7];
  g.gnw = (const float*)d_in[base+8];  g.gnb = (const float*)d_in[base+9];
  g.gnms= (const float*)d_in[base+10];
  return g;
}

static void run_gin_block(hipStream_t stream,
    float* x, int R, const int* src, const int* dst, int ne,
    const float* eattr, const int* gb, const float* cnt, int G,
    const GinParams& P, const __bf16* wsw1, const __bf16* wsw2,
    float* h0, float* H1, float* H2,
    float* bn_s, float* bn_q, float* bn_a, float* bn_b2,
    float* gsum, float* gsq, float* gmean, float* ginv,
    float* gmf, float* gvf, int last_act) {
  size_t n128 = (size_t)R * 128;
  size_t nGF  = (size_t)G * 128;
  copy_f32<<<(unsigned)((n128 + 255) / 256), 256, 0, stream>>>(x, h0, n128);
  {
    size_t tot = (size_t)ne * 32;
    gin_messages<<<(unsigned)((tot + 255) / 256), 256, 0, stream>>>(x, eattr, src, dst, ne, h0);
  }
  zero_f32<<<1, 256, 0, stream>>>(bn_s, 256);
  zero_f32<<<1, 256, 0, stream>>>(bn_q, 256);
  unsigned gtiles = (unsigned)((R + 15) / 16);
  gemm_wmma<<<gtiles, 128, 0, stream>>>(h0, R, 128, 256, wsw1, P.b1, nullptr, nullptr, 0, H1);
  bn_stats<<<(R + 255) / 256, 256, 0, stream>>>(H1, R, bn_s, bn_q);
  bn_finalize<<<1, 256, 0, stream>>>(bn_s, bn_q, P.bng, P.bnb, R, bn_a, bn_b2);
  gemm_wmma<<<gtiles, 128, 0, stream>>>(H1, R, 256, 128, wsw2, P.b2, bn_a, bn_b2, 1, H2);
  zero_f32<<<(G + 255) / 256, 256, 0, stream>>>(gsum, G);
  zero_f32<<<(G + 255) / 256, 256, 0, stream>>>(gsq, G);
  ln_rowstats<<<R, 128, 0, stream>>>(H2, gb, R, gsum, gsq);
  ln_finalize<<<(G + 255) / 256, 256, 0, stream>>>(gsum, gsq, cnt, G, gmean, ginv);
  zero_f32<<<(unsigned)((nGF + 255) / 256), 256, 0, stream>>>(gmf, nGF);
  ln_apply<<<(unsigned)((n128 + 255) / 256), 256, 0, stream>>>(H2, gb, R, gmean, ginv, P.lnw, P.lnb, gmf);
  gn_div<<<(unsigned)((nGF + 255) / 256), 256, 0, stream>>>(gmf, cnt, G);
  zero_f32<<<(unsigned)((nGF + 255) / 256), 256, 0, stream>>>(gvf, nGF);
  gn_apply1<<<(unsigned)((n128 + 255) / 256), 256, 0, stream>>>(H2, gb, R, gmf, P.gnms, gvf);
  gn_finalize<<<(unsigned)((nGF + 255) / 256), 256, 0, stream>>>(gvf, cnt, G);
  gn_apply2<<<(unsigned)((n128 + 255) / 256), 256, 0, stream>>>(H2, gb, R, gvf, P.gnw, P.gnb, last_act, x);
}

extern "C" void kernel_launch(void* const* d_in, const int* in_sizes, int n_in,
                              void* d_out, int out_size, void* d_ws, size_t ws_size,
                              hipStream_t stream) {
  (void)n_in; (void)ws_size;
  const int N = in_sizes[0] / 9;
  const int E = in_sizes[3] / 3;
  const int A = in_sizes[6];
  const int G = out_size / 128;
  const int Rmax = (N > E) ? N : E;

  const int*   x_idx  = (const int*)d_in[0];
  const float* pos_g  = (const float*)d_in[1];
  const float* pos_ex = (const float*)d_in[2];
  const int*   eattr_i= (const int*)d_in[3];
  const float* blg    = (const float*)d_in[4];
  const float* ble    = (const float*)d_in[5];
  const float* bag    = (const float*)d_in[6];
  const float* bae    = (const float*)d_in[7];
  const int*   eidx   = (const int*)d_in[8];    // src=eidx[0:E], dst=eidx[E:2E]
  const int*   bbidx  = (const int*)d_in[9];
  const int*   batch  = (const int*)d_in[10];
  const int*   ebatch = (const int*)d_in[11];

  // params (recursive insertion order, base index 13)
  Tab9 at; for (int i = 0; i < 9; ++i) at.t[i] = (const float*)d_in[13 + i];
  Tab3 bt; for (int i = 0; i < 3; ++i) bt.t[i] = (const float*)d_in[22 + i];
  const float* distW = (const float*)d_in[25]; const float* distB = (const float*)d_in[26];
  const float* pAW   = (const float*)d_in[27]; const float* pAB   = (const float*)d_in[28];
  const float* lenW  = (const float*)d_in[29]; const float* lenB  = (const float*)d_in[30];
  const float* pLW   = (const float*)d_in[31]; const float* pLB   = (const float*)d_in[32];
  const float* angW  = (const float*)d_in[33]; const float* angB  = (const float*)d_in[34];
  const float* pGW   = (const float*)d_in[35]; const float* pGB   = (const float*)d_in[36];
  GinParams agin[3], bgin[3];
  for (int i = 0; i < 3; ++i) { agin[i] = getGin(d_in, 37 + i * 11); bgin[i] = getGin(d_in, 70 + i * 11); }
  const float* flnW = (const float*)d_in[103];
  const float* flnB = (const float*)d_in[104];

  // workspace carve-up
  char* wp = (char*)d_ws;
  auto carve = [&](size_t bytes) -> char* {
    char* r = wp; wp += (bytes + 255) & ~(size_t)255; return r;
  };
  float* atom_x = (float*)carve((size_t)N * 128 * 4);
  float* edge_x = (float*)carve((size_t)E * 128 * 4);
  float* angle_x= (float*)carve((size_t)A * 128 * 4);
  float* h0     = (float*)carve((size_t)Rmax * 128 * 4);
  float* H1     = (float*)carve((size_t)Rmax * 256 * 4);
  float* H2     = (float*)carve((size_t)Rmax * 128 * 4);
  float* bn_s   = (float*)carve(256 * 4);
  float* bn_q   = (float*)carve(256 * 4);
  float* bn_a   = (float*)carve(256 * 4);
  float* bn_b2  = (float*)carve(256 * 4);
  float* gsum   = (float*)carve((size_t)G * 4);
  float* gsq    = (float*)carve((size_t)G * 4);
  float* gmean  = (float*)carve((size_t)G * 4);
  float* ginv   = (float*)carve((size_t)G * 4);
  float* gmf    = (float*)carve((size_t)G * 128 * 4);
  float* gvf    = (float*)carve((size_t)G * 128 * 4);
  float* cntA   = (float*)carve((size_t)G * 4);
  float* cntE   = (float*)carve((size_t)G * 4);
  float* pool   = (float*)carve((size_t)G * 128 * 4);
  __bf16* wsw   = (__bf16*)carve((size_t)6 * 65536 * 2);

  // counts
  zero_f32<<<(G + 255) / 256, 256, 0, stream>>>(cntA, G);
  zero_f32<<<(G + 255) / 256, 256, 0, stream>>>(cntE, G);
  count_batch<<<(N + 255) / 256, 256, 0, stream>>>(batch, N, cntA);
  count_batch<<<(E + 255) / 256, 256, 0, stream>>>(ebatch, E, cntE);

  // swizzle all GIN weights into WMMA B-fragment order (bf16)
  for (int i = 0; i < 3; ++i) {
    __bf16* wb = wsw + (size_t)i * 65536;        // bond block i
    __bf16* wa = wsw + (size_t)(3 + i) * 65536;  // atom block i
    swizzle_w<<<128, 256, 0, stream>>>(bgin[i].W1, 128, 256, wb);
    swizzle_w<<<128, 256, 0, stream>>>(bgin[i].W2, 256, 128, wb + 32768);
    swizzle_w<<<128, 256, 0, stream>>>(agin[i].W1, 128, 256, wa);
    swizzle_w<<<128, 256, 0, stream>>>(agin[i].W2, 256, 128, wa + 32768);
  }

  // encoders
  enc_atom <<<N, 128, 0, stream>>>(at, x_idx, pos_g, pos_ex, distW, distB, pAW, pAB, N, atom_x);
  enc_bond <<<E, 128, 0, stream>>>(bt, eattr_i, blg, ble, lenW, lenB, pLW, pLB, E, edge_x);
  enc_angle<<<A, 128, 0, stream>>>(bag, bae, angW, angB, pGW, pGB, A, angle_x);

  // 3 x (bond GIN, atom GIN)
  for (int i = 0; i < 3; ++i) {
    int last_act = (i < 2) ? 1 : 0;
    __bf16* wb = wsw + (size_t)i * 65536;
    __bf16* wa = wsw + (size_t)(3 + i) * 65536;
    run_gin_block(stream, edge_x, E, bbidx, bbidx + A, A, angle_x, ebatch, cntE, G,
                  bgin[i], wb, wb + 32768, h0, H1, H2, bn_s, bn_q, bn_a, bn_b2,
                  gsum, gsq, gmean, ginv, gmf, gvf, last_act);
    run_gin_block(stream, atom_x, N, eidx, eidx + E, E, edge_x, batch, cntA, G,
                  agin[i], wa, wa + 32768, h0, H1, H2, bn_s, bn_q, bn_a, bn_b2,
                  gsum, gsq, gmean, ginv, gmf, gvf, last_act);
  }

  // mean pool + final LayerNorm
  size_t nGF = (size_t)G * 128;
  zero_f32<<<(unsigned)((nGF + 255) / 256), 256, 0, stream>>>(pool, nGF);
  pool_add<<<(unsigned)(((size_t)N * 128 + 255) / 256), 256, 0, stream>>>(atom_x, batch, N, pool);
  final_ln<<<G, 128, 0, stream>>>(pool, cntA, flnW, flnB, G, (float*)d_out);
}
